// CWS_Model_62749472195400
// MI455X (gfx1250) — matrix operand order repaired and checked by reading, not compiled, not run on hardware
//
#include <hip/hip_runtime.h>
#include <hip/hip_bf16.h>

// ---------------------------------------------------------------------------
// CWS segmental-LSTM forward for MI455X (gfx1250, wave32).
// All GEMMs run on v_wmma_f32_16x16x32_bf16 (f32 accumulate).
// Weights are pre-packed once per call into the WMMA B layout:
//   Bpack[k/2][n] = dword{bf16(W[k][n]), bf16(W[k+1][n])}
// A matrices are staged in LDS as packed bf16 pairs and loaded per the ISA
// 16-bit A layout (lanes0-15: K0-7/K16-23, lanes16-31: K8-15/K24-31).
// State history uses ring buffers of depth 5 (L=4 history + 1 write slot) so
// a step never writes the slot another workgroup may still read.
// Nonlinearities use v_rcp_f32-based fast sigmoid/tanh (the scan is latency
// bound; the IEEE divide expansion costs ~8 VALU ops per element).
// ---------------------------------------------------------------------------

#define BB  128
#define TT  256
#define LL  4
#define DCC 128
#define DWW 128
#define HH  256

typedef __attribute__((ext_vector_type(16))) __bf16 v16bf;
typedef __attribute__((ext_vector_type(8)))  float  v8f;

union Frag16 { v16bf v; unsigned u[8]; };

__device__ __forceinline__ unsigned pack2bf(float a, float b){
  union { __bf16 h[2]; unsigned u; } c;
  c.h[0] = (__bf16)a; c.h[1] = (__bf16)b;
  return c.u;
}
__device__ __forceinline__ float bfhalf(unsigned u, int hi){
  union { unsigned x; float f; } c;
  c.x = hi ? (u & 0xffff0000u) : (u << 16);
  return c.f;
}
// fast transcendentals: v_exp_f32 + v_rcp_f32, no IEEE divide expansion
__device__ __forceinline__ float sigm(float x){
  return __builtin_amdgcn_rcpf(1.0f + __expf(-x));
}
__device__ __forceinline__ float tanh_fast(float x){
  // tanh(x) = 1 - 2/(1+exp(2x)); exact at both saturation ends via rcp(inf)=0
  return 1.0f - 2.0f * __builtin_amdgcn_rcpf(1.0f + __expf(2.0f * x));
}
// A fragment from LDS (packed bf16-pair dwords, row-major, strideDw dwords/row)
__device__ __forceinline__ v16bf ldsA(const unsigned* s, int strideDw, int rowBase,
                                      int kDw, int lane){
  int m = lane & 15, hi = lane >> 4;
  const unsigned* p = s + (rowBase + m) * strideDw + kDw + hi * 4;
  Frag16 f;
#pragma unroll
  for (int i = 0; i < 4; ++i){ f.u[i] = p[i]; f.u[4 + i] = p[8 + i]; }
  return f.v;
}
// B fragment from global pre-packed weights (Bpack[k/2][n], Nw dwords/row)
__device__ __forceinline__ v16bf glbB(const unsigned* g, int Nw, int colBase,
                                      int kDw, int lane){
  int n = lane & 15, hi = lane >> 4;
  const unsigned* p = g + (size_t)(kDw + hi * 4) * Nw + colBase + n;
  Frag16 f;
#pragma unroll
  for (int i = 0; i < 8; ++i) f.u[i] = p[(size_t)(i + (i >> 2) * 4) * Nw];
  return f.v;
}
__device__ __forceinline__ v8f wmma32(v16bf a, v16bf b, v8f c){
  return __builtin_amdgcn_wmma_f32_16x16x32_bf16(false, a, false, b,
                                                 (short)0, c, false, false);
}

// --------------------------- weight pre-pack --------------------------------
__global__ __launch_bounds__(256) void k_prep(
    const float* __restrict__ reset_W, const float* __restrict__ com_W,
    const float* __restrict__ lstm_kernel, const float* __restrict__ pred_W,
    unsigned* __restrict__ resetWp, unsigned* __restrict__ comWp,
    unsigned* __restrict__ lstmWp, unsigned* __restrict__ predWp)
{
  const int NR = 4 * 64 * 128, NC = 64 * 128, NL = 192 * 1024, NP = 128 * 128;
  const int tot = NR + NC + NL + NP;
  for (int idx = blockIdx.x * blockDim.x + threadIdx.x; idx < tot;
       idx += gridDim.x * blockDim.x){
    if (idx < NR){
      int w = idx / 8192, rem = idx % 8192, kd = rem >> 7, n2 = rem & 127;
      const float* base = reset_W + (size_t)w * 128 * 128;
      resetWp[idx] = pack2bf(base[(2 * kd) * 128 + n2], base[(2 * kd + 1) * 128 + n2]);
    } else if (idx < NR + NC){
      int i2 = idx - NR, kd = i2 >> 7, n2 = i2 & 127;
      comWp[i2] = pack2bf(com_W[(2 * kd) * 128 + n2], com_W[(2 * kd + 1) * 128 + n2]);
    } else if (idx < NR + NC + NL){
      int i2 = idx - NR - NC, kd = i2 >> 10, n2 = i2 & 1023;
      lstmWp[i2] = pack2bf(lstm_kernel[(size_t)(2 * kd) * 1024 + n2],
                           lstm_kernel[(size_t)(2 * kd + 1) * 1024 + n2]);
    } else {
      int i2 = idx - NR - NC - NL, kd = i2 >> 7, n2 = i2 & 127;
      predWp[i2] = pack2bf(pred_W[(2 * kd) * 128 + n2], pred_W[(2 * kd + 1) * 128 + n2]);
    }
  }
}

// ------------------------------ init step -----------------------------------
// bos step: h0=c0=0 => z = bos @ K[0:128,:] + bias, identical across batch.
__global__ __launch_bounds__(256) void k_init(
    const float* __restrict__ bos, const float* __restrict__ lstm_kernel,
    const float* __restrict__ lstm_bias, const float* __restrict__ pred_W,
    const float* __restrict__ pred_b,
    float* __restrict__ ring_pred, float* __restrict__ ring_c,
    float* __restrict__ ring_h)
{
  __shared__ float z0[4 * HH];
  __shared__ float c1s[HH], h1s[HH], p0s[DWW];
  const int tid = threadIdx.x;
  for (int col = tid; col < 4 * HH; col += 256){
    float acc = lstm_bias[col];
    for (int k = 0; k < DWW; ++k)
      acc += bos[k] * lstm_kernel[(size_t)k * 4 * HH + col];
    z0[col] = acc;
  }
  __syncthreads();
  if (tid < HH){
    float i = z0[tid], j = z0[HH + tid], o = z0[3 * HH + tid];
    float nc = sigm(i) * tanh_fast(j);      // c0 = 0 so the f-gate term vanishes
    float nh = tanh_fast(nc) * sigm(o);
    c1s[tid] = nc; h1s[tid] = nh;
  }
  __syncthreads();
  if (tid < DWW){
    float acc = pred_b[tid];
    for (int k = 0; k < HH; ++k) acc += h1s[k] * pred_W[(size_t)k * DWW + tid];
    p0s[tid] = tanh_fast(acc);
  }
  __syncthreads();
  for (int idx = tid; idx < 5 * BB * HH; idx += 256){
    int h = idx % HH;
    ring_c[idx] = c1s[h];
    ring_h[idx] = h1s[h];
  }
  for (int idx = tid; idx < 5 * BB * DWW; idx += 256)
    ring_pred[idx] = p0s[idx % DWW];
}

// ----------------------- per-step stage 1: word + score ---------------------
// grid = 8 batch-tiles x 4 word-lengths; block = 256 (8 waves).
__global__ __launch_bounds__(256) void k_word(
    const int* __restrict__ chars, const float* __restrict__ char_emb,
    const float* __restrict__ reset_b, const float* __restrict__ com_b,
    const float* __restrict__ score_U,
    const unsigned* __restrict__ resetWp, const unsigned* __restrict__ comWp,
    const float* __restrict__ ring_pred,
    float* __restrict__ word_out, float* __restrict__ score_out, int t)
{
  __shared__ unsigned embP[64 * 64];   // 64 rows (16b x 4c) x 64 packed dwords
  __shared__ unsigned apP [64 * 64];   // g*emb, same layout
  __shared__ float sscore[16];
  const int tid = threadIdx.x;
  const int lane = tid & 31, wv = tid >> 5;
  const int bt = blockIdx.x & 7, w = blockIdx.x >> 3;
  if (tid < 16) sscore[tid] = 0.0f;
  // gather embeddings of the L=4 window chars for 16 batch rows
  for (int idx = tid; idx < 64 * 64; idx += 256){
    int row = idx >> 6, d = idx & 63;
    int bl = row >> 2, c = row & 3;
    int b = bt * 16 + bl;
    int cid = (t >= c) ? chars[b * TT + t - c] : 0;   // front padding -> id 0
    const float* e = char_emb + (size_t)cid * DCC + 2 * d;
    embP[idx] = pack2bf(e[0], e[1]);
  }
  __syncthreads();
  const int n = lane & 15, hi = lane >> 4;
  // ---- reset-gate GEMM (64x128x128) + sigmoid * emb -> apP ----
  for (int p = 0; p < 4; ++p){
    int tileIdx = wv * 4 + p;
    int mt = tileIdx >> 3, nt = tileIdx & 7;
    v8f acc = {};
#pragma unroll
    for (int kt = 0; kt < 4; ++kt){
      v16bf a  = ldsA(embP, 64, mt * 16, kt * 16, lane);
      v16bf bf = glbB(resetWp + w * 64 * 128, 128, nt * 16, kt * 16, lane);
      acc = wmma32(a, bf, acc);
    }
    int col = nt * 16 + n;
#pragma unroll
    for (int r = 0; r < 8; ++r){
      int rr = mt * 16 + r + hi * 8;                  // C layout: M=r / r+8
      float g  = sigm(acc[r] + reset_b[w * DCC + col]);
      float ev = bfhalf(embP[rr * 64 + (col >> 1)], col & 1);
      float val = g * ev;
      float other = __shfl_xor(val, 1);               // pair adjacent columns
      if ((lane & 1) == 0)
        apP[rr * 64 + (col >> 1)] = pack2bf(val, other);
    }
  }
  __syncthreads();
  // ---- compose GEMM (64x128x128) + tanh + masked mean + score ----
  const float invlen = 1.0f / (float)(w + 1);
  const int slotRd = (t + 4 - w) % 5;                 // == (t-1-w) mod 5
  for (int p = 0; p < 4; ++p){
    int tileIdx = wv * 4 + p;
    int mt = tileIdx >> 3, nt = tileIdx & 7;
    v8f acc = {};
#pragma unroll
    for (int kt = 0; kt < 4; ++kt){
      v16bf a  = ldsA(apP, 64, mt * 16, kt * 16, lane);
      v16bf bf = glbB(comWp, 128, nt * 16, kt * 16, lane);
      acc = wmma32(a, bf, acc);
    }
    int col = nt * 16 + n;
    float pv[8];
#pragma unroll
    for (int r = 0; r < 8; ++r) pv[r] = tanh_fast(acc[r] + com_b[col]);
    float s0 = 0.0f, s1 = 0.0f;
#pragma unroll
    for (int c = 0; c < 4; ++c)
      if (c <= w){ s0 += pv[c]; s1 += pv[4 + c]; }    // mean over first w+1 chars
    s0 *= invlen; s1 *= invlen;
    int bl0 = mt * 4 + hi * 2, bl1 = bl0 + 1;
    int b0 = bt * 16 + bl0, b1 = bt * 16 + bl1;
    word_out[((size_t)b0 * LL + w) * DWW + col] = s0;
    word_out[((size_t)b1 * LL + w) * DWW + col] = s1;
    float u = score_U[col];
    float p0 = (ring_pred[(size_t)slotRd * BB * DWW + (size_t)b0 * DWW + col] + u) * s0;
    float p1 = (ring_pred[(size_t)slotRd * BB * DWW + (size_t)b1 * DWW + col] + u) * s1;
    atomicAdd(&sscore[bl0], p0);
    atomicAdd(&sscore[bl1], p1);
  }
  __syncthreads();
  if (tid < 16)
    score_out[(bt * 16 + tid) * LL + w] = sscore[tid];
}

// ------------------- per-step stage 2: argmax + LSTM gates ------------------
// grid = 8 M-tiles x 16 h-column tiles; block = 32 (one wave).
// Each wave computes 4 accumulators (i,j,f,o slices) over K = 384.
__global__ __launch_bounds__(32) void k_lstm(
    const float* __restrict__ score, const float* __restrict__ word,
    float* __restrict__ ring_c, float* __restrict__ ring_h,
    const unsigned* __restrict__ lstmWp, const float* __restrict__ lstm_bias,
    float* __restrict__ out, int t)
{
  __shared__ unsigned Ap[16 * 192];   // A = [word_b(128) | h_prev(256)] bf16 pairs
  __shared__ int bestS[16];
  const int tid = threadIdx.x;
  const int mt = blockIdx.x & 7, ncol = blockIdx.x >> 3;
  const int wmax = (t < 3) ? t : 3;   // reference masks wlens <= t+1
  if (tid < 16){
    int b = mt * 16 + tid;
    float bs = score[b * LL];
    int best = 0;
    for (int w2 = 1; w2 <= wmax; ++w2){
      float s = score[b * LL + w2];
      if (s > bs){ bs = s; best = w2; }
    }
    bestS[tid] = best;
    if (ncol == 0){
      out[(size_t)b * TT + t] = bs;                              // scores.T
      out[(size_t)BB * TT + (size_t)b * TT + t] = (float)(best + 1); // wl.T
    }
  }
  __syncthreads();
  for (int idx = tid; idx < 16 * 192; idx += 32){
    int row = idx / 192, d = idx % 192;
    int b = mt * 16 + row;
    int best = bestS[row];
    float f0, f1;
    if (d < 64){
      const float* p = word + ((size_t)b * LL + best) * DWW + 2 * d;
      f0 = p[0]; f1 = p[1];
    } else {
      int k = (d - 64) * 2;
      int slotRd = (t + 4 - best) % 5;
      const float* p = ring_h + (size_t)slotRd * BB * HH + (size_t)b * HH + k;
      f0 = p[0]; f1 = p[1];
    }
    Ap[idx] = pack2bf(f0, f1);
  }
  __syncthreads();
  const int lane = tid, n = lane & 15, hi = lane >> 4;
  v8f acc[4] = {};
  for (int kt = 0; kt < 12; ++kt){
    // prefetch next K-step's B rows into L2/L0 (global_prefetch_b8)
    if (kt + 1 < 12){
      const unsigned* pf = lstmWp + (size_t)((kt + 1) * 16 + (hi ? 8 : 0)) * 1024
                                  + ncol * 16 + n;
      __builtin_prefetch(pf, 0, 3);
    }
    v16bf a = ldsA(Ap, 192, 0, kt * 16, lane);
#pragma unroll
    for (int g4 = 0; g4 < 4; ++g4){
      v16bf bf = glbB(lstmWp, 1024, g4 * 256 + ncol * 16, kt * 16, lane);
      acc[g4] = wmma32(a, bf, acc[g4]);
    }
  }
  const int hcol = ncol * 16 + n;
  const int wrSlot = t % 5;
#pragma unroll
  for (int r = 0; r < 8; ++r){
    int row = r + hi * 8;
    int b = mt * 16 + row;
    int best = bestS[row];
    int slotRd = (t + 4 - best) % 5;
    float iv = acc[0][r] + lstm_bias[hcol];
    float jv = acc[1][r] + lstm_bias[HH + hcol];
    float fv = acc[2][r] + lstm_bias[2 * HH + hcol];
    float ov = acc[3][r] + lstm_bias[3 * HH + hcol];
    float cprev = ring_c[(size_t)slotRd * BB * HH + (size_t)b * HH + hcol];
    float nc = cprev * sigm(fv) + sigm(iv) * tanh_fast(jv);
    float nh = tanh_fast(nc) * sigm(ov);
    ring_c[(size_t)wrSlot * BB * HH + (size_t)b * HH + hcol] = nc;
    ring_h[(size_t)wrSlot * BB * HH + (size_t)b * HH + hcol] = nh;
  }
}

// ------------------------ per-step stage 3: pred ----------------------------
// grid = 8 M-tiles; block = 256 (wave wv owns N-tile wv). GEMM 128x128x256.
__global__ __launch_bounds__(256) void k_pred(
    const float* __restrict__ ring_h, const unsigned* __restrict__ predWp,
    const float* __restrict__ pred_b, float* __restrict__ ring_pred, int t)
{
  __shared__ unsigned Ap[16 * 128];
  const int tid = threadIdx.x, lane = tid & 31, wv = tid >> 5;
  const int mt = blockIdx.x;
  const int slot = t % 5;
  for (int idx = tid; idx < 16 * 128; idx += 256){
    int row = idx >> 7, d = idx & 127;
    int b = mt * 16 + row;
    const float* p = ring_h + (size_t)slot * BB * HH + (size_t)b * HH + 2 * d;
    Ap[idx] = pack2bf(p[0], p[1]);
  }
  __syncthreads();
  const int n = lane & 15, hi = lane >> 4;
  v8f acc = {};
  for (int kt = 0; kt < 8; ++kt){
    v16bf a  = ldsA(Ap, 128, 0, kt * 16, lane);
    v16bf bf = glbB(predWp, 128, wv * 16, kt * 16, lane);
    acc = wmma32(a, bf, acc);
  }
  int col = wv * 16 + n;
#pragma unroll
  for (int r = 0; r < 8; ++r){
    int b = mt * 16 + r + hi * 8;
    ring_pred[(size_t)slot * BB * DWW + (size_t)b * DWW + col] =
        tanh_fast(acc[r] + pred_b[col]);
  }
}

// ------------------------------- launcher -----------------------------------
extern "C" void kernel_launch(void* const* d_in, const int* in_sizes, int n_in,
                              void* d_out, int out_size, void* d_ws, size_t ws_size,
                              hipStream_t stream)
{
  const int*   chars       = (const int*)  d_in[0];
  const float* char_emb    = (const float*)d_in[1];
  const float* reset_W     = (const float*)d_in[2];
  const float* reset_b     = (const float*)d_in[3];
  const float* com_W       = (const float*)d_in[4];
  const float* com_b       = (const float*)d_in[5];
  const float* lstm_kernel = (const float*)d_in[6];
  const float* lstm_bias   = (const float*)d_in[7];
  const float* pred_W      = (const float*)d_in[8];
  const float* pred_b      = (const float*)d_in[9];
  const float* score_U     = (const float*)d_in[10];
  const float* bos         = (const float*)d_in[11];
  float* out = (float*)d_out;

  char* ws = (char*)d_ws;
  size_t off = 0;
  auto alloc = [&](size_t dwords) -> void* {
    void* p = ws + off;
    off += dwords * 4;
    off = (off + 255) & ~(size_t)255;
    return p;
  };
  unsigned* resetWp = (unsigned*)alloc(4 * 64 * 128);
  unsigned* comWp   = (unsigned*)alloc(64 * 128);
  unsigned* lstmWp  = (unsigned*)alloc(192 * 1024);
  unsigned* predWp  = (unsigned*)alloc(128 * 128);
  float* word       = (float*)alloc((size_t)BB * LL * DWW);
  float* scoreBuf   = (float*)alloc((size_t)BB * LL);
  float* ring_pred  = (float*)alloc((size_t)5 * BB * DWW);
  float* ring_c     = (float*)alloc((size_t)5 * BB * HH);
  float* ring_h     = (float*)alloc((size_t)5 * BB * HH);

  hipLaunchKernelGGL(k_prep, dim3(256), dim3(256), 0, stream,
                     reset_W, com_W, lstm_kernel, pred_W,
                     resetWp, comWp, lstmWp, predWp);
  hipLaunchKernelGGL(k_init, dim3(1), dim3(256), 0, stream,
                     bos, lstm_kernel, lstm_bias, pred_W, pred_b,
                     ring_pred, ring_c, ring_h);
  for (int t = 0; t < TT; ++t){
    hipLaunchKernelGGL(k_word, dim3(32), dim3(256), 0, stream,
                       chars, char_emb, reset_b, com_b, score_U,
                       resetWp, comWp, ring_pred, word, scoreBuf, t);
    hipLaunchKernelGGL(k_lstm, dim3(128), dim3(32), 0, stream,
                       scoreBuf, word, ring_c, ring_h, lstmWp, lstm_bias, out, t);
    hipLaunchKernelGGL(k_pred, dim3(8), dim3(256), 0, stream,
                       ring_h, predWp, pred_b, ring_pred, t);
  }
}
